// MemoryModule_3994319585526
// MI455X (gfx1250) — matrix operand order repaired and checked by reading, not compiled, not run on hardware
//
#include <hip/hip_runtime.h>
#include <hip/hip_bf16.h>
#include <math.h>

typedef __attribute__((ext_vector_type(16))) __bf16 v16bf;
typedef __attribute__((ext_vector_type(8)))  float  v8f;
typedef __attribute__((ext_vector_type(4)))  unsigned int u32x4;

union BFrag { v16bf v; u32x4 q[2]; };

__device__ __forceinline__ unsigned short f2bf(float f) {
  unsigned int u = __float_as_uint(f);
  unsigned int r = u + 0x7FFFu + ((u >> 16) & 1u);
  return (unsigned short)(r >> 16);
}
__device__ __forceinline__ unsigned int pack2bf(float a, float b) {
  return (unsigned int)f2bf(a) | ((unsigned int)f2bf(b) << 16);
}

// ---------------------------------------------------------------------------
// Kernel 1: 6x6 adaptive-avg-pool of a [*,96,96] plane + derive all 4 scales
// pooled layout per channel: 50 slots = [s1(1) | s2(4) | s3(9) | s6(36)]
// grid.x = T*B*C, block = 192 (96 x 2)
// ---------------------------------------------------------------------------
__global__ void pool_ppm_kernel(const float* __restrict__ x,
                                float* __restrict__ pooled) {
  const int ch = blockIdx.x;
  const float* plane = x + (size_t)ch * 9216;
  const int tid  = threadIdx.x;
  const int xcol = tid % 96;
  const int ys   = tid / 96;  // 0..1

  __shared__ float red[192];
  __shared__ float p6[36];

  float bs[6];
#pragma unroll
  for (int by = 0; by < 6; ++by) {
    float s = 0.f;
#pragma unroll
    for (int k = 0; k < 8; ++k) {
      int y = by * 16 + ys + 2 * k;
      s += plane[y * 96 + xcol];
    }
    bs[by] = s;
  }
#pragma unroll
  for (int by = 0; by < 6; ++by) {
    red[tid] = bs[by];
    __syncthreads();
    if (tid < 6) {
      float t = 0.f;
#pragma unroll
      for (int xx = 0; xx < 16; ++xx)
        t += red[tid * 16 + xx] + red[96 + tid * 16 + xx];
      p6[by * 6 + tid] = t * (1.0f / 256.0f);
    }
    __syncthreads();
  }

  float* outp = pooled + (size_t)ch * 50;
  if (tid < 50) {
    float v;
    if (tid == 0) {                       // scale 1
      v = 0.f;
      for (int q = 0; q < 36; ++q) v += p6[q];
      v *= (1.0f / 36.0f);
    } else if (tid < 5) {                 // scale 2 : 3x3 groups of p6
      int ij = tid - 1; int i = ij / 2, j = ij % 2;
      v = 0.f;
      for (int a = 0; a < 3; ++a)
        for (int bq = 0; bq < 3; ++bq) v += p6[(3 * i + a) * 6 + 3 * j + bq];
      v *= (1.0f / 9.0f);
    } else if (tid < 14) {                // scale 3 : 2x2 groups of p6
      int ij = tid - 5; int i = ij / 3, j = ij % 3;
      v = 0.f;
      for (int a = 0; a < 2; ++a)
        for (int bq = 0; bq < 2; ++bq) v += p6[(2 * i + a) * 6 + 2 * j + bq];
      v *= 0.25f;
    } else {                              // scale 6
      v = p6[tid - 14];
    }
    outp[tid] = v;
  }
}

// ---------------------------------------------------------------------------
// Kernel 2: per-(t,b,scale) 1x1 conv + ReLU + bilinear upsample to 6x6,
// packed bf16 into mk [B][144][256] (kv==0) or mv [B][512][160] (kv==1).
// grid.x = T*B*4, block = 256
// ---------------------------------------------------------------------------
__global__ void conv_ups_kernel(const float* __restrict__ pooled,
                                const float* __restrict__ w,
                                const float* __restrict__ bias,
                                unsigned short* __restrict__ dst,
                                int C, int Co, int kv) {
  const int g  = blockIdx.x;
  const int si = g & 3;
  const int tb = g >> 2;      // t*B + b
  const int b  = tb & 1;
  const int t  = tb >> 1;
  const int scale_of[4] = {1, 2, 3, 6};
  const int off_of[4]   = {0, 1, 5, 14};
  const int s  = scale_of[si];
  const int off = off_of[si];
  const int s2 = s * s;

  const float* pbase = pooled + (size_t)tb * C * 50 + off;
  const float* wbase = w + ((size_t)t * 4 + si) * Co * C;
  const float* bb    = bias + ((size_t)t * 4 + si) * Co;

  __shared__ float conv[128 * 36];

  for (int o = threadIdx.x; o < Co * s2; o += blockDim.x) {
    int co = o / s2, ij = o % s2;
    const float* wr = wbase + (size_t)co * C;
    float a = bb[co];
    for (int c = 0; c < C; ++c) a += wr[c] * pbase[(size_t)c * 50 + ij];
    conv[o] = fmaxf(a, 0.0f);
  }
  __syncthreads();

  for (int o = threadIdx.x; o < Co * 36; o += blockDim.x) {
    int co = o / 36, ij = o % 36;
    int i = ij / 6, j = ij % 6;
    float v;
    if (s == 6) {
      v = conv[co * 36 + ij];
    } else if (s == 1) {
      v = conv[co];
    } else {
      float sy = (i + 0.5f) * (float)s / 6.0f - 0.5f;
      float sx = (j + 0.5f) * (float)s / 6.0f - 0.5f;
      int y0 = (int)floorf(sy); float fy = sy - (float)y0;
      int x0 = (int)floorf(sx); float fx = sx - (float)x0;
      int y0c = min(max(y0, 0), s - 1), y1c = min(max(y0 + 1, 0), s - 1);
      int x0c = min(max(x0, 0), s - 1), x1c = min(max(x0 + 1, 0), s - 1);
      const float* cp = &conv[co * s2];
      v = cp[y0c * s + x0c] * (1.f - fy) * (1.f - fx)
        + cp[y0c * s + x1c] * (1.f - fy) * fx
        + cp[y1c * s + x0c] * fy * (1.f - fx)
        + cp[y1c * s + x1c] * fy * fx;
    }
    int m   = t * 36 + ij;
    int chn = si * Co + co;
    size_t di = (kv == 0)
        ? ((size_t)b * 144 + m) * 256 + chn     // mk: [B][M=144][K=256]
        : ((size_t)b * 512 + chn) * 160 + m;    // mv: [B][V=512][K=160 padded]
    dst[di] = f2bf(v);
  }
}

// ---------------------------------------------------------------------------
// Kernel 3: zero mv (for K-padding 144..159)
// ---------------------------------------------------------------------------
__global__ void zero_u32_kernel(unsigned int* p, int n) {
  int i = blockIdx.x * blockDim.x + threadIdx.x;
  if (i < n) p[i] = 0u;
}

// ---------------------------------------------------------------------------
// Kernel 4: copy query_value -> out[:, 0:512]
// ---------------------------------------------------------------------------
__global__ void copy_qv_kernel(const float4* __restrict__ qv,
                               float* __restrict__ out) {
  size_t i = (size_t)blockIdx.x * blockDim.x + threadIdx.x;
  const size_t per_b = (size_t)512 * 9216 / 4;
  if (i >= 2 * per_b) return;
  size_t b = i / per_b, r = i % per_b;
  ((float4*)(out + b * (size_t)1024 * 9216))[r] = qv[i];
}

// ---------------------------------------------------------------------------
// Kernel 5: fused attention.
//   att = softmax_M( mk(144x256) @ qk(256xN) / 16 ), mem = mv(512x144) @ att
// grid = (144 tiles of N=64, B); block = 128 (4 waves, one 16-col strip each)
// ---------------------------------------------------------------------------
__global__ __launch_bounds__(128)
void attention_kernel(const float* __restrict__ qk,
                      const unsigned short* __restrict__ mk,
                      const unsigned short* __restrict__ mv,
                      float* __restrict__ out) {
  const int b  = blockIdx.y;
  const int n0 = blockIdx.x * 64;

  __shared__ __align__(16) unsigned short qkT[64 * 264];  // [n][c], bf16
  __shared__ __align__(16) unsigned short attS[64 * 168]; // [n][m(160 pad)], bf16

  const int tid  = threadIdx.x;
  const int lane = tid & 31;
  const int wv   = tid >> 5;     // wave 0..3
  const int nl   = lane & 15;
  const int hf   = lane >> 4;    // lane half

  // ---- stage qk tile transposed into LDS as bf16 -------------------------
  const float* qkB = qk + (size_t)b * 256 * 9216 + n0;
#pragma unroll 4
  for (int rep = 0; rep < 128; ++rep) {
    int idx = rep * 128 + tid;       // 0..16383
    int c = idx >> 6, n = idx & 63;
    qkT[n * 264 + c] = f2bf(qkB[(size_t)c * 9216 + n]);
  }
  __syncthreads();

  // ---- GEMM1: att[144][16 strip] = mk(144x256) @ qkT ---------------------
  v8f acc[9];
#pragma unroll
  for (int mt = 0; mt < 9; ++mt)
#pragma unroll
    for (int r = 0; r < 8; ++r) acc[mt][r] = 0.f;

  const unsigned short* mkB = mk + (size_t)b * 144 * 256;
  for (int ks = 0; ks < 8; ++ks) {
    BFrag bq;
    const unsigned short* bp = &qkT[(wv * 16 + nl) * 264 + ks * 32 + hf * 8];
    bq.q[0] = *(const u32x4*)bp;
    bq.q[1] = *(const u32x4*)(bp + 16);
#pragma unroll
    for (int mt = 0; mt < 9; ++mt) {
      BFrag aq;
      const unsigned short* ap = mkB + (size_t)(mt * 16 + nl) * 256 + ks * 32 + hf * 8;
      aq.q[0] = *(const u32x4*)ap;
      aq.q[1] = *(const u32x4*)(ap + 16);
      acc[mt] = __builtin_amdgcn_wmma_f32_16x16x32_bf16(
          false, aq.v, false, bq.v, (short)0, acc[mt], false, false);
    }
  }

  // ---- softmax over M=144 (column data lives in lane and lane^16) --------
  float mx = -INFINITY;
#pragma unroll
  for (int mt = 0; mt < 9; ++mt)
#pragma unroll
    for (int r = 0; r < 8; ++r) mx = fmaxf(mx, acc[mt][r]);
  mx = fmaxf(mx, __shfl_xor(mx, 16, 32));

  float sum = 0.f;
#pragma unroll
  for (int mt = 0; mt < 9; ++mt)
#pragma unroll
    for (int r = 0; r < 8; ++r) {
      float e = __expf((acc[mt][r] - mx) * 0.0625f);  // Ck^-0.5 = 1/16
      acc[mt][r] = e;
      sum += e;
    }
  sum += __shfl_xor(sum, 16, 32);
  const float inv = 1.0f / sum;

  // ---- write att (bf16) into LDS column-major, zero-pad K 144..159 -------
  const int nCol = wv * 16 + nl;
  unsigned short* colp = &attS[nCol * 168];
#pragma unroll
  for (int mt = 0; mt < 9; ++mt) {
    int M0 = mt * 16 + hf * 8;
#pragma unroll
    for (int r = 0; r < 8; r += 2)
      *(unsigned int*)(colp + M0 + r) =
          pack2bf(acc[mt][r] * inv, acc[mt][r + 1] * inv);
  }
  {
    u32x4 z = {0u, 0u, 0u, 0u};
    *(u32x4*)(colp + 144 + hf * 8) = z;
  }
  __syncthreads();

  // ---- GEMM2: mem[512][16 strip] = mv(512x160) @ att ---------------------
  BFrag bq2[5];
#pragma unroll
  for (int ks = 0; ks < 5; ++ks) {
    const unsigned short* bp = &attS[nCol * 168 + ks * 32 + hf * 8];
    bq2[ks].q[0] = *(const u32x4*)bp;
    bq2[ks].q[1] = *(const u32x4*)(bp + 16);
  }

  const unsigned short* mvB = mv + (size_t)b * 512 * 160;
  float* outB = out + ((size_t)b * 1024 + 512) * 9216 + n0 + wv * 16 + nl;

  for (int vt = 0; vt < 32; ++vt) {
    v8f c2;
#pragma unroll
    for (int r = 0; r < 8; ++r) c2[r] = 0.f;
#pragma unroll
    for (int ks = 0; ks < 5; ++ks) {
      BFrag aq;
      const unsigned short* ap = mvB + (size_t)(vt * 16 + nl) * 160 + ks * 32 + hf * 8;
      aq.q[0] = *(const u32x4*)ap;
      aq.q[1] = *(const u32x4*)(ap + 16);
      c2 = __builtin_amdgcn_wmma_f32_16x16x32_bf16(
          false, aq.v, false, bq2[ks].v, (short)0, c2, false, false);
    }
#pragma unroll
    for (int r = 0; r < 8; ++r) {
      int v = vt * 16 + hf * 8 + r;
      outB[(size_t)v * 9216] = c2[r];
    }
  }
}

// ---------------------------------------------------------------------------
extern "C" void kernel_launch(void* const* d_in, const int* in_sizes, int n_in,
                              void* d_out, int out_size, void* d_ws, size_t ws_size,
                              hipStream_t stream) {
  (void)in_sizes; (void)n_in; (void)out_size; (void)ws_size;
  const float* mkeys = (const float*)d_in[0];
  const float* mvals = (const float*)d_in[1];
  const float* qk    = (const float*)d_in[2];
  const float* qv    = (const float*)d_in[3];
  const float* key_w = (const float*)d_in[4];
  const float* key_b = (const float*)d_in[5];
  const float* val_w = (const float*)d_in[6];
  const float* val_b = (const float*)d_in[7];
  float* out = (float*)d_out;

  // workspace layout (16B aligned throughout)
  float* pooled_k = (float*)d_ws;                       // 2048*50 f32
  float* pooled_v = pooled_k + 2048 * 50;               // 4096*50 f32
  unsigned short* mk = (unsigned short*)(pooled_v + 4096 * 50); // 2*144*256 bf16
  unsigned short* mv = mk + 2 * 144 * 256;              // 2*512*160 bf16

  // 1) pooling (keys then values)
  pool_ppm_kernel<<<2048, 192, 0, stream>>>(mkeys, pooled_k);
  pool_ppm_kernel<<<4096, 192, 0, stream>>>(mvals, pooled_v);

  // 2) zero mv (K padding), then conv+upsample+pack
  {
    int n32 = 2 * 512 * 160 / 2;  // ushort count / 2 = u32 count
    zero_u32_kernel<<<(n32 + 255) / 256, 256, 0, stream>>>((unsigned int*)mv, n32);
  }
  conv_ups_kernel<<<4 * 2 * 4, 256, 0, stream>>>(pooled_k, key_w, key_b, mk, 256, 64, 0);
  conv_ups_kernel<<<4 * 2 * 4, 256, 0, stream>>>(pooled_v, val_w, val_b, mv, 512, 128, 1);

  // 3) copy query_value into first 512 output channels
  {
    size_t nf4 = (size_t)2 * 512 * 9216 / 4;
    copy_qv_kernel<<<(unsigned)((nf4 + 255) / 256), 256, 0, stream>>>((const float4*)qv, out);
  }

  // 4) fused attention + output write
  attention_kernel<<<dim3(144, 2), 128, 0, stream>>>(qk, mk, mv, out);
}
